// TransformerBlock_56667798503727
// MI455X (gfx1250) — matrix operand order repaired and checked by reading, not compiled, not run on hardware
//
#include <hip/hip_runtime.h>

typedef __attribute__((ext_vector_type(16))) __bf16 v16bf;
typedef __attribute__((ext_vector_type(8)))  float  v8f;

#if __has_builtin(__builtin_amdgcn_global_load_async_to_lds_b128) && \
    __has_builtin(__builtin_amdgcn_s_wait_asynccnt)
#define HAVE_ASYNC_LDS 1
// parameter types discovered via compile probe: (v4i AS1*, v4i AS3*, imm, imm)
typedef int v4i_async __attribute__((vector_size(16)));
typedef __attribute__((address_space(1))) v4i_async* async_gptr;
typedef __attribute__((address_space(3))) v4i_async* async_lptr;
#define ASYNC_COPY_B128(gsrc, ldst) \
  __builtin_amdgcn_global_load_async_to_lds_b128((async_gptr)(gsrc), (async_lptr)(ldst), 0, 0)
#else
#define HAVE_ASYNC_LDS 0
#endif

// trivially-constructible 16B vector (union-friendly, 16B aligned)
struct __attribute__((aligned(16))) U4 { unsigned int x, y, z, w; };

// fragment view: 16 bf16 = 8 dwords = 2 x 16B
union VF {
  v16bf f;
  U4 q[2];
  unsigned int u[8];
  unsigned short s[16];
};

__device__ __forceinline__ unsigned short f2bfu(float f) {
  return __builtin_bit_cast(unsigned short, (__bf16)f);
}

// interleave two 8x-bf16 rows into k-pair order: [a0,b0,a1,b1,...]
__device__ __forceinline__ void ilv(U4 a, U4 b, U4& lo, U4& hi) {
  union { U4 v; unsigned short s[8]; } ua, ub, ol, oh;
  ua.v = a; ub.v = b;
#pragma unroll
  for (int j = 0; j < 4; ++j) { ol.s[2*j] = ua.s[j];   ol.s[2*j+1] = ub.s[j]; }
#pragma unroll
  for (int j = 0; j < 4; ++j) { oh.s[2*j] = ua.s[j+4]; oh.s[2*j+1] = ub.s[j+4]; }
  lo = ol.v; hi = oh.v;
}

// ---------------------------------------------------------------- cvt f32->bf16
__global__ void cvt_bf16_kernel(const float* __restrict__ in, __bf16* __restrict__ out,
                                long long n) {
  long long i = (long long)blockIdx.x * blockDim.x + threadIdx.x;
  long long stride = (long long)gridDim.x * blockDim.x;
  for (; i < n; i += stride) out[i] = (__bf16)in[i];
}

// ---------------------------------------------------------------- layernorm -> bf16
__global__ __launch_bounds__(256) void layernorm_bf16_kernel(
    const float* __restrict__ x, const float* __restrict__ g,
    const float* __restrict__ be, __bf16* __restrict__ out) {
  const int D = 1024;
  __shared__ float red[256];
  int row = blockIdx.x, tid = threadIdx.x;
  const float* xr = x + (size_t)row * D;
  float4 v = ((const float4*)xr)[tid];
  red[tid] = v.x + v.y + v.z + v.w;
  __syncthreads();
  for (int off = 128; off > 0; off >>= 1) {
    if (tid < off) red[tid] += red[tid + off];
    __syncthreads();
  }
  float mu = red[0] * (1.0f / D);
  __syncthreads();
  float dx0 = v.x - mu, dx1 = v.y - mu, dx2 = v.z - mu, dx3 = v.w - mu;
  red[tid] = dx0 * dx0 + dx1 * dx1 + dx2 * dx2 + dx3 * dx3;
  __syncthreads();
  for (int off = 128; off > 0; off >>= 1) {
    if (tid < off) red[tid] += red[tid + off];
    __syncthreads();
  }
  float rstd = rsqrtf(red[0] * (1.0f / D) + 1e-5f);
  int c = tid * 4;
  float4 gv = ((const float4*)g)[tid];
  float4 bv = ((const float4*)be)[tid];
  __bf16* orow = out + (size_t)row * D + c;
  orow[0] = (__bf16)(dx0 * rstd * gv.x + bv.x);
  orow[1] = (__bf16)(dx1 * rstd * gv.y + bv.y);
  orow[2] = (__bf16)(dx2 * rstd * gv.z + bv.z);
  orow[3] = (__bf16)(dx3 * rstd * gv.w + bv.w);
}

// ---------------------------------------------------------------- WMMA GEMM
// C[M,N] = act(A @ B + bias) (+resid). Block tile 128x128, 256 threads = 8 waves.
// Wave (wm,wn): rows wm*32..+31 (two A frags), cols wn*64..+63 (4 B frags) -> 8 WMMAs/K-step.
#define FLAG_GELU 1
#define FLAG_QKV  2

__global__ __launch_bounds__(256) void gemm_wmma_kernel(
    const __bf16* __restrict__ A, const __bf16* __restrict__ Bw,
    const float* __restrict__ bias, const float* __restrict__ resid,
    float* __restrict__ outF, __bf16* __restrict__ outB,
    int M, int N, int K, int flags) {
  const int TK = 32, TN = 128;
  __shared__ unsigned short As[128 * 32];      // A tile row-major      (8 KB)
  __shared__ unsigned short Bp[16 * 128 * 2];  // B tile [k/2][col][2]  (8 KB)
  int tid = threadIdx.x, lane = tid & 31, wave = tid >> 5;
  int hf = lane >> 4, l15 = lane & 15;
  int wm = wave & 3, wn = wave >> 2;
  int mblk = blockIdx.y * 128, nblk = blockIdx.x * TN;
  int bkp = tid >> 4, bcg = tid & 15;  // B staging: kpair / col-group

  v8f acc[2][4];
#pragma unroll
  for (int a = 0; a < 2; ++a)
#pragma unroll
    for (int f = 0; f < 4; ++f)
#pragma unroll
      for (int e = 0; e < 8; ++e) acc[a][f][e] = 0.0f;

  for (int kk = 0; kk < K; kk += TK) {
    // ---- stage A tile: 128x32 = 4096 elems, 16 per thread (two b128 async copies)
    {
      int eid = tid * 16;
#pragma unroll
      for (int j = 0; j < 2; ++j) {
        int e2 = eid + j * 8, ar = e2 >> 5, ac = e2 & 31;
        const __bf16* agp = A + (size_t)(mblk + ar) * K + kk + ac;
#if HAVE_ASYNC_LDS
        ASYNC_COPY_B128(agp, &As[e2]);
#else
        *(U4*)&As[e2] = *(const U4*)agp;
#endif
      }
    }
    // ---- stage B tile pair-interleaved in registers (wide stores)
    {
      const __bf16* b0 = Bw + (size_t)(kk + 2 * bkp) * N + nblk + bcg * 8;
      U4 r0 = *(const U4*)b0;
      U4 r1 = *(const U4*)(b0 + N);
      U4 lo, hi;
      ilv(r0, r1, lo, hi);
      *(U4*)&Bp[bkp * 256 + bcg * 16] = lo;
      *(U4*)&Bp[bkp * 256 + bcg * 16 + 8] = hi;
    }
    if (kk + TK < K)  // gfx1250 global_prefetch_b8
      __builtin_prefetch(A + (size_t)(mblk + (tid >> 2)) * K + kk + TK + (tid & 3) * 8, 0, 1);
#if HAVE_ASYNC_LDS
    __builtin_amdgcn_s_wait_asynccnt(0);
#endif
    __syncthreads();

    // A fragments: two contiguous 16B k-runs per lane, two m-subtiles
    VF af[2];
#pragma unroll
    for (int a = 0; a < 2; ++a) {
      int arow = wm * 32 + a * 16 + l15;
      af[a].q[0] = *(const U4*)&As[arow * 32 + 16 * hf];
      af[a].q[1] = *(const U4*)&As[arow * 32 + 16 * hf + 8];
    }
#pragma unroll
    for (int f = 0; f < 4; ++f) {
      VF bfr;
      int col = wn * 64 + f * 16 + l15;
#pragma unroll
      for (int i = 0; i < 8; ++i)  // one b32 per VGPR slot (pairs merge to 2addr)
        bfr.u[i] = *(const unsigned int*)&Bp[(8 * hf + i) * 256 + col * 2];
#pragma unroll
      for (int a = 0; a < 2; ++a)  // B fragment reused from registers
        acc[a][f] = __builtin_amdgcn_wmma_f32_16x16x32_bf16(
            false, af[a].f, false, bfr.f, (short)0, acc[a][f], false, false);
    }
    __syncthreads();
  }

  // epilogue: C row = slot + 8*hf, col = l15
#pragma unroll
  for (int f = 0; f < 4; ++f) {
    int n = nblk + wn * 64 + f * 16 + l15;
    float bv = bias[n];
#pragma unroll
    for (int a = 0; a < 2; ++a) {
#pragma unroll
      for (int r = 0; r < 8; ++r) {
        int m = mblk + wm * 32 + a * 16 + r + 8 * hf;
        float v = acc[a][f][r] + bv;
        if (flags & FLAG_GELU)
          v = 0.5f * v * (1.0f + erff(v * 0.70710678118654752f));
        if (resid) v += resid[(size_t)m * N + n];
        if (outF) outF[(size_t)m * N + n] = v;
        if (outB) {
          size_t idx;
          if (flags & FLAG_QKV) {  // (B,S,D) -> (B,H,S,dk)
            int b = m >> 11, s = m & 2047, h = n >> 6, d = n & 63;
            idx = (((size_t)b * 16 + h) * 2048 + s) * 64 + d;
          } else {
            idx = (size_t)m * N + n;
          }
          outB[idx] = (__bf16)v;
        }
      }
    }
  }
}

// ---------------------------------------------------------------- flash attention
// Q,K,V: (B*H, S, 64) bf16. ctx: (B,S,D) bf16. 128 threads = 4 independent waves.
__global__ __launch_bounds__(128) void flash_attn_kernel(
    const __bf16* __restrict__ Q, const __bf16* __restrict__ Kp,
    const __bf16* __restrict__ Vp_, const int* __restrict__ amask,
    __bf16* __restrict__ ctx) {
  const int S = 2048, H = 16, DK = 64, D = 1024;
  const float NEG = -1.0e9f, MINIT = -1.0e30f;
  __shared__ unsigned short Ks[4][32 * 64];       // per-wave K chunk row-major
  __shared__ unsigned short Vq[4][16 * 64 * 2];   // per-wave V [k/2][d][2]
  __shared__ unsigned short Ps[4][16 * 32];       // per-wave P transpose buf

  int tid = threadIdx.x, lane = tid & 31, w = tid >> 5;
  int hf = lane >> 4, l15 = lane & 15;
  int bh = blockIdx.y;
  int b = bh >> 4, h = bh & (H - 1);
  int qbase = blockIdx.x * 64 + w * 16;

  const unsigned short* qus = (const unsigned short*)(Q + (size_t)bh * S * DK);
  const unsigned short* kus = (const unsigned short*)(Kp + (size_t)bh * S * DK);
  const unsigned short* vus = (const unsigned short*)(Vp_ + (size_t)bh * S * DK);

  // Q fragments (two 16x32 d-chunks) resident in registers
  VF qf[2];
  int qrow = qbase + l15;
#pragma unroll
  for (int c = 0; c < 2; ++c) {
    qf[c].q[0] = *(const U4*)(qus + (size_t)qrow * DK + 32 * c + 16 * hf);
    qf[c].q[1] = *(const U4*)(qus + (size_t)qrow * DK + 32 * c + 16 * hf + 8);
  }

  v8f o[4];
  float m_[8], l_[8], alpha[8], p0[8], p1[8];
#pragma unroll
  for (int f = 0; f < 4; ++f)
#pragma unroll
    for (int e = 0; e < 8; ++e) o[f][e] = 0.0f;
#pragma unroll
  for (int r = 0; r < 8; ++r) { m_[r] = MINIT; l_[r] = 0.0f; }

  U4 z4 = {0u, 0u, 0u, 0u};
  int nch = (qbase + 16 + 31) >> 5;  // causal: need keys <= qbase+15
  for (int ch = 0; ch < nch; ++ch) {
    int kc = ch * 32;
    // ---- stage K chunk (row-major straight copy: async if available)
    {
      int krow = kc + lane;
      const unsigned short* ksr = kus + (size_t)krow * DK;
#pragma unroll
      for (int it = 0; it < 8; ++it) {
#if HAVE_ASYNC_LDS
        if (krow < S)
          ASYNC_COPY_B128(ksr + it * 8, &Ks[w][lane * 64 + it * 8]);
        else
          *(U4*)&Ks[w][lane * 64 + it * 8] = z4;
#else
        U4 kd = (krow < S) ? *(const U4*)(ksr + it * 8) : z4;
        *(U4*)&Ks[w][lane * 64 + it * 8] = kd;
#endif
      }
    }
    // ---- stage V chunk pair-interleaved: lane = (kpair, d-half)
    {
      int kp = lane & 15, dh = lane >> 4;
      int k0 = kc + 2 * kp;
#pragma unroll
      for (int j = 0; j < 4; ++j) {
        U4 r0 = (k0 < S) ? *(const U4*)(vus + (size_t)k0 * DK + dh * 32 + j * 8) : z4;
        U4 r1 = (k0 + 1 < S) ? *(const U4*)(vus + (size_t)(k0 + 1) * DK + dh * 32 + j * 8) : z4;
        U4 lo, hi;
        ilv(r0, r1, lo, hi);
        *(U4*)&Vq[w][kp * 128 + dh * 64 + j * 16] = lo;
        *(U4*)&Vq[w][kp * 128 + dh * 64 + j * 16 + 8] = hi;
      }
    }
#if HAVE_ASYNC_LDS
    __builtin_amdgcn_s_wait_asynccnt(0);
#endif
    __builtin_amdgcn_wave_barrier();

    // ---- scores: 4 WMMAs (2 key subtiles x 2 d-chunks), contiguous 32B K reads
    v8f s0, s1;
#pragma unroll
    for (int e = 0; e < 8; ++e) { s0[e] = 0.0f; s1[e] = 0.0f; }
#pragma unroll
    for (int c = 0; c < 2; ++c) {
      VF bk0, bk1;
      const unsigned short* kb0 = &Ks[w][l15 * 64 + 32 * c + 16 * hf];
      const unsigned short* kb1 = &Ks[w][(16 + l15) * 64 + 32 * c + 16 * hf];
      bk0.q[0] = *(const U4*)kb0; bk0.q[1] = *(const U4*)(kb0 + 8);
      bk1.q[0] = *(const U4*)kb1; bk1.q[1] = *(const U4*)(kb1 + 8);
      s0 = __builtin_amdgcn_wmma_f32_16x16x32_bf16(false, qf[c].f, false, bk0.f,
                                                   (short)0, s0, false, false);
      s1 = __builtin_amdgcn_wmma_f32_16x16x32_bf16(false, qf[c].f, false, bk1.f,
                                                   (short)0, s1, false, false);
    }

    // ---- mask + online softmax
    int col0 = kc + l15, col1 = kc + 16 + l15;
    int pm0 = (col0 < S) ? amask[b * S + col0] : 0;
    int pm1 = (col1 < S) ? amask[b * S + col1] : 0;
#pragma unroll
    for (int r = 0; r < 8; ++r) {
      int row = qbase + r + 8 * hf;
      float v0 = s0[r] * 0.125f;
      float v1 = s1[r] * 0.125f;
      if (col0 > row || pm0 == 0) v0 = NEG;
      if (col1 > row || pm1 == 0) v1 = NEG;
      float vm = fmaxf(v0, v1);
      for (int off = 8; off > 0; off >>= 1)
        vm = fmaxf(vm, __shfl_xor(vm, off, 16));
      float mnew = fmaxf(m_[r], vm);
      float e0 = (v0 > 0.5f * NEG) ? __expf(v0 - mnew) : 0.0f;
      float e1 = (v1 > 0.5f * NEG) ? __expf(v1 - mnew) : 0.0f;
      float rs = e0 + e1;
      for (int off = 8; off > 0; off >>= 1) rs += __shfl_xor(rs, off, 16);
      float al = (m_[r] > 0.5f * MINIT) ? __expf(m_[r] - mnew) : 0.0f;
      l_[r] = l_[r] * al + rs;
      m_[r] = mnew;
      alpha[r] = al;
      p0[r] = e0;
      p1[r] = e1;
    }
#pragma unroll
    for (int f = 0; f < 4; ++f)
#pragma unroll
      for (int r = 0; r < 8; ++r) o[f][r] *= alpha[r];

    // ---- transpose P through LDS (C layout -> A-operand layout)
#pragma unroll
    for (int r = 0; r < 8; ++r) {
      int prow = r + 8 * hf;
      Ps[w][prow * 32 + l15] = f2bfu(p0[r]);
      Ps[w][prow * 32 + 16 + l15] = f2bfu(p1[r]);
    }
    __builtin_amdgcn_wave_barrier();
    VF pf;
    pf.q[0] = *(const U4*)&Ps[w][l15 * 32 + 16 * hf];
    pf.q[1] = *(const U4*)&Ps[w][l15 * 32 + 16 * hf + 8];

    // ---- O += P @ V : 4 WMMAs, one b32 per V slot
#pragma unroll
    for (int f = 0; f < 4; ++f) {
      VF vfr;
      int dcol = f * 16 + l15;
#pragma unroll
      for (int i = 0; i < 8; ++i)
        vfr.u[i] = *(const unsigned int*)&Vq[w][(8 * hf + i) * 128 + dcol * 2];
      o[f] = __builtin_amdgcn_wmma_f32_16x16x32_bf16(false, pf.f, false, vfr.f,
                                                     (short)0, o[f], false, false);
    }
    __builtin_amdgcn_wave_barrier();
  }

  // ---- write ctx; fully-masked rows -> 0 (matches nan_to_num)
#pragma unroll
  for (int f = 0; f < 4; ++f)
#pragma unroll
    for (int r = 0; r < 8; ++r) {
      int q = qbase + r + 8 * hf;
      float li = l_[r];
      float val = (li > 0.0f) ? o[f][r] / li : 0.0f;
      ctx[((size_t)b * S + q) * D + h * DK + f * 16 + l15] = (__bf16)val;
    }
}

// ---------------------------------------------------------------- launch
extern "C" void kernel_launch(void* const* d_in, const int* in_sizes, int n_in,
                              void* d_out, int out_size, void* d_ws, size_t ws_size,
                              hipStream_t stream) {
  (void)in_sizes; (void)n_in; (void)out_size; (void)ws_size;
  const float* x   = (const float*)d_in[0];
  const int* amask = (const int*)d_in[1];
  const float* Wq  = (const float*)d_in[2];
  const float* bq  = (const float*)d_in[3];
  const float* Wk  = (const float*)d_in[4];
  const float* bk  = (const float*)d_in[5];
  const float* Wv  = (const float*)d_in[6];
  const float* bv  = (const float*)d_in[7];
  const float* Wo  = (const float*)d_in[8];
  const float* bo  = (const float*)d_in[9];
  const float* W1  = (const float*)d_in[10];
  const float* b1  = (const float*)d_in[11];
  const float* W2  = (const float*)d_in[12];
  const float* b2  = (const float*)d_in[13];
  const float* g1  = (const float*)d_in[14];
  const float* be1 = (const float*)d_in[15];
  const float* g2  = (const float*)d_in[16];
  const float* be2 = (const float*)d_in[17];
  float* out = (float*)d_out;

  const size_t MB = 1ull << 20;
  char* ws = (char*)d_ws;
  __bf16* wq_bf = (__bf16*)(ws + 0 * MB);
  __bf16* wk_bf = (__bf16*)(ws + 2 * MB);
  __bf16* wv_bf = (__bf16*)(ws + 4 * MB);
  __bf16* wo_bf = (__bf16*)(ws + 6 * MB);
  __bf16* w1_bf = (__bf16*)(ws + 8 * MB);
  __bf16* w2_bf = (__bf16*)(ws + 16 * MB);
  __bf16* hbuf  = (__bf16*)(ws + 24 * MB);
  __bf16* qb    = (__bf16*)(ws + 40 * MB);
  __bf16* kb    = (__bf16*)(ws + 56 * MB);
  __bf16* vb    = (__bf16*)(ws + 72 * MB);
  __bf16* ctxb  = (__bf16*)(ws + 88 * MB);
  __bf16* ffn1b = qb;  // reuse q/k/v/ctx region after attention

  const int Mrows = 8192;  // B*S
  auto cvt = [&](const float* src, __bf16* dst, long long n) {
    int blocks = (int)((n + 255) / 256);
    cvt_bf16_kernel<<<blocks, 256, 0, stream>>>(src, dst, n);
  };
  cvt(Wq, wq_bf, 1024LL * 1024);
  cvt(Wk, wk_bf, 1024LL * 1024);
  cvt(Wv, wv_bf, 1024LL * 1024);
  cvt(Wo, wo_bf, 1024LL * 1024);
  cvt(W1, w1_bf, 1024LL * 4096);
  cvt(W2, w2_bf, 4096LL * 1024);

  layernorm_bf16_kernel<<<Mrows, 256, 0, stream>>>(x, g1, be1, hbuf);

  dim3 g1024(1024 / 128, Mrows / 128);
  gemm_wmma_kernel<<<g1024, 256, 0, stream>>>(hbuf, wq_bf, bq, nullptr, nullptr,
                                              qb, Mrows, 1024, 1024, FLAG_QKV);
  gemm_wmma_kernel<<<g1024, 256, 0, stream>>>(hbuf, wk_bf, bk, nullptr, nullptr,
                                              kb, Mrows, 1024, 1024, FLAG_QKV);
  gemm_wmma_kernel<<<g1024, 256, 0, stream>>>(hbuf, wv_bf, bv, nullptr, nullptr,
                                              vb, Mrows, 1024, 1024, FLAG_QKV);

  flash_attn_kernel<<<dim3(2048 / 64, 4 * 16), 128, 0, stream>>>(qb, kb, vb,
                                                                 amask, ctxb);

  gemm_wmma_kernel<<<g1024, 256, 0, stream>>>(ctxb, wo_bf, bo, x, out, nullptr,
                                              Mrows, 1024, 1024, 0);

  layernorm_bf16_kernel<<<Mrows, 256, 0, stream>>>(out, g2, be2, hbuf);

  dim3 g4096(4096 / 128, Mrows / 128);
  gemm_wmma_kernel<<<g4096, 256, 0, stream>>>(hbuf, w1_bf, b1, nullptr, nullptr,
                                              ffn1b, Mrows, 4096, 1024, FLAG_GELU);

  gemm_wmma_kernel<<<g1024, 256, 0, stream>>>(ffn1b, w2_bf, b2, out, out, nullptr,
                                              Mrows, 1024, 4096, 0);
}
